// NLinear_38474317038419
// MI455X (gfx1250) — compile-verified, matrix operand order checked by reading, not run on hardware
//
#include <hip/hip_runtime.h>

typedef __attribute__((ext_vector_type(16))) __bf16 v16bf;
typedef __attribute__((ext_vector_type(8)))  __bf16 v8bf;
typedef __attribute__((ext_vector_type(4)))  __bf16 v4bf;
typedef __attribute__((ext_vector_type(2)))  __bf16 v2bf;
typedef __attribute__((ext_vector_type(8)))  float  v8f;

#define BATCH 1024
#define NMAT  128
#define DIN   256
#define DOUT  256

#define BM 64     // batch rows per block
#define BN 128    // output cols per block
#define KC 64     // K-chunk staged in LDS
#define LDS_STRIDE 72  // bf16 elems per row; 144 B = 9 * 16 B (aligned, conflict-free b128)

// Load a 16x32 bf16 fragment row for this lane from an LDS row.
// ISA layout (16-bit A/B): lanes 0-15 hold K = kk+0..7 (v8) and kk+16..23 (v8);
// lanes 16-31 hold K = kk+8..15 and kk+24..31.
__device__ __forceinline__ v16bf load_frag(const __bf16* row, int kk, int khalf) {
    v8bf f0 = *(const v8bf*)(row + kk + khalf * 8);
    v8bf f1 = *(const v8bf*)(row + kk + 16 + khalf * 8);
    return __builtin_shufflevector(f0, f1, 0,1,2,3,4,5,6,7,8,9,10,11,12,13,14,15);
}

__global__ __launch_bounds__(256)
void nlinear_wmma_bf16x3(const float* __restrict__ x,
                         const float* __restrict__ w,
                         const float* __restrict__ bias,
                         float* __restrict__ out)
{
    // hi/lo bf16 planes: A = x tile [m][k], B = w tile transposed [o][k]
    __shared__ __align__(16) __bf16 As[2][BM][LDS_STRIDE];
    __shared__ __align__(16) __bf16 Bs[2][BN][LDS_STRIDE];

    const int n  = blockIdx.z;
    const int b0 = blockIdx.x * BM;
    const int c0 = blockIdx.y * BN;

    const int tid   = threadIdx.x;
    const int lane  = tid & 31;
    const int wv    = tid >> 5;        // wave 0..7
    const int lrow  = lane & 15;
    const int khalf = lane >> 4;
    const int m0    = (wv & 3) * 16;   // wave's M-tile (4 tiles cover BM=64)
    const int ng    = (wv >> 2) * 64;  // wave's N-group (2 groups of 4 tiles cover BN=128)

    v8f acc[4] = {};

    // A loader: 16 rows/pass, 16 threads * float4 per row
    const int a_tr = tid >> 4;          // 0..15
    const int a_tc = (tid & 15) * 4;    // 0..60
    // B loader: 2(k) x 4(o) micro-tile per thread per pass (bank-friendly transpose)
    const int b_oc = (tid >> 3) * 4;    // 0..124
    const int b_k2 = (tid & 7) * 2;     // 0..14

    for (int kc = 0; kc < DIN; kc += KC) {
        // ---- stage x tile (BM x KC), split f32 -> bf16 hi/lo ----
        #pragma unroll
        for (int p = 0; p < 4; ++p) {
            const int row = p * 16 + a_tr;
            const float4 v = *(const float4*)(x + ((size_t)(b0 + row) * NMAT + n) * DIN + kc + a_tc);
            const float vv[4] = {v.x, v.y, v.z, v.w};
            v4bf h, l;
            #pragma unroll
            for (int j = 0; j < 4; ++j) {
                __bf16 hb = (__bf16)vv[j];
                h[j] = hb;
                l[j] = (__bf16)(vv[j] - (float)hb);
            }
            *(v4bf*)&As[0][row][a_tc] = h;
            *(v4bf*)&As[1][row][a_tc] = l;
        }
        // ---- stage w tile (KC x BN), split + transpose into [o][k] ----
        #pragma unroll
        for (int p = 0; p < 4; ++p) {
            const int k = p * 16 + b_k2;
            const float* wp = w + ((size_t)n * DIN + (kc + k)) * DOUT + c0 + b_oc;
            const float4 r0 = *(const float4*)(wp);
            const float4 r1 = *(const float4*)(wp + DOUT);
            const float a0[4] = {r0.x, r0.y, r0.z, r0.w};
            const float a1[4] = {r1.x, r1.y, r1.z, r1.w};
            #pragma unroll
            for (int j = 0; j < 4; ++j) {
                __bf16 h0 = (__bf16)a0[j];
                __bf16 h1 = (__bf16)a1[j];
                v2bf hp = {h0, h1};
                v2bf lp = {(__bf16)(a0[j] - (float)h0), (__bf16)(a1[j] - (float)h1)};
                *(v2bf*)&Bs[0][b_oc + j][k] = hp;
                *(v2bf*)&Bs[1][b_oc + j][k] = lp;
            }
        }
        __syncthreads();

        // ---- compute: 2 kk-steps x 4 N-tiles x 3 WMMAs (hi*hi, lo*hi, hi*lo) ----
        #pragma unroll
        for (int kk = 0; kk < KC; kk += 32) {
            const v16bf a_hi = load_frag(&As[0][m0 + lrow][0], kk, khalf);
            const v16bf a_lo = load_frag(&As[1][m0 + lrow][0], kk, khalf);
            #pragma unroll
            for (int t = 0; t < 4; ++t) {
                const v16bf b_hi = load_frag(&Bs[0][ng + t * 16 + lrow][0], kk, khalf);
                const v16bf b_lo = load_frag(&Bs[1][ng + t * 16 + lrow][0], kk, khalf);
                acc[t] = __builtin_amdgcn_wmma_f32_16x16x32_bf16(
                             false, a_hi, false, b_hi, (short)0, acc[t], false, false);
                acc[t] = __builtin_amdgcn_wmma_f32_16x16x32_bf16(
                             false, a_lo, false, b_hi, (short)0, acc[t], false, false);
                acc[t] = __builtin_amdgcn_wmma_f32_16x16x32_bf16(
                             false, a_hi, false, b_lo, (short)0, acc[t], false, false);
            }
        }
        __syncthreads();
    }

    // ---- epilogue: + bias, store. Each lane's column is fixed per tile. ----
    #pragma unroll
    for (int t = 0; t < 4; ++t) {
        const int o  = c0 + ng + t * 16 + lrow;
        const float bv = bias[n * DOUT + o];
        #pragma unroll
        for (int r = 0; r < 8; ++r) {
            const int m = b0 + m0 + khalf * 8 + r;
            out[((size_t)m * NMAT + n) * DOUT + o] = acc[t][r] + bv;
        }
    }
}

extern "C" void kernel_launch(void* const* d_in, const int* in_sizes, int n_in,
                              void* d_out, int out_size, void* d_ws, size_t ws_size,
                              hipStream_t stream) {
    const float* x  = (const float*)d_in[0];
    const float* w  = (const float*)d_in[1];
    const float* bs = (const float*)d_in[2];
    float* out = (float*)d_out;
    (void)in_sizes; (void)n_in; (void)out_size; (void)d_ws; (void)ws_size;

    dim3 grid(BATCH / BM, DOUT / BN, NMAT);  // (16, 2, 128)
    dim3 block(256);
    hipLaunchKernelGGL(nlinear_wmma_bf16x3, grid, block, 0, stream, x, w, bs, out);
}
